// GraphConvLayer_2413771620668
// MI455X (gfx1250) — compile-verified, matrix operand order checked by reading, not run on hardware
//
#include <hip/hip_runtime.h>
#include <hip/hip_bf16.h>

#define NNODES 50000
#define NEDGES 800000
#define DD 64
#define NREL 8

typedef __attribute__((ext_vector_type(16))) _Float16 v16h;
typedef __attribute__((ext_vector_type(8)))  float    v8f;

// ---------------------------------------------------------------------------
// Kernel 1: zero the aggregation buffer (lives in d_out) and the counts.
// ---------------------------------------------------------------------------
__global__ void gcl_zero(float* __restrict__ agg, float* __restrict__ cnt) {
    int idx = blockIdx.x * blockDim.x + threadIdx.x;
    if (idx < NNODES * DD) agg[idx] = 0.0f;
    if (idx < NNODES)      cnt[idx] = 0.0f;
}

// ---------------------------------------------------------------------------
// Kernel 2: w[r][j] = sum_k W_r[r][j][k]   (8*128 outputs, 64-wide sums)
// ---------------------------------------------------------------------------
__global__ void gcl_wsum(const float* __restrict__ W_r, float* __restrict__ w) {
    int idx = blockIdx.x * blockDim.x + threadIdx.x;
    if (idx >= NREL * 2 * DD) return;
    const float4* p = (const float4*)(W_r + (size_t)idx * DD);
    float s = 0.0f;
#pragma unroll
    for (int i = 0; i < 16; ++i) {
        float4 v = p[i];
        s += v.x + v.y + v.z + v.w;
    }
    w[idx] = s;
}

// ---------------------------------------------------------------------------
// Kernel 2b: pre-pack W_lin (64x128 f32, row-major) into f16 B-fragments in
// the exact per-lane WMMA layout: packB[((kc*4 + t)*32 + lane)*16 + i].
// Lane holds col n = t*16 + (lane&15), k-half hs = lane>>4:
//   i in [0,8):  k = kc*32 + hs*8 + i
//   i in [8,16): k = kc*32 + 16 + hs*8 + (i-8)
// One thread per fragment-lane (4*4*32 = 512 threads), 32 contiguous bytes out.
// ---------------------------------------------------------------------------
__global__ void gcl_packB(const float* __restrict__ Wlin, _Float16* __restrict__ packB) {
    int idx = blockIdx.x * blockDim.x + threadIdx.x;   // 0..511
    if (idx >= 4 * 4 * 32) return;
    int lane = idx & 31;
    int t    = (idx >> 5) & 3;
    int kc   = idx >> 7;
    int n    = t * 16 + (lane & 15);
    int hs   = lane >> 4;
    const float* wrow = Wlin + (size_t)n * (2 * DD) + kc * 32 + hs * 8;
    _Float16* outp = packB + (size_t)idx * 16;
#pragma unroll
    for (int i = 0; i < 8; ++i)  outp[i]     = (_Float16)wrow[i];
#pragma unroll
    for (int i = 0; i < 8; ++i)  outp[8 + i] = (_Float16)wrow[16 + i];
}

// ---------------------------------------------------------------------------
// Kernel 3: edge pass. One wave32 per edge; lane owns 2 of the 64 features.
// logit = <x[dst], w_dst[rel]> + <x[src], w_src[rel]>; gate = sigmoid(logit);
// atomically accumulate x[src]*gate into agg[dst], and 1.0 into cnt[dst].
// x (12.8 MB) and agg (12.8 MB) both live in the 192 MB L2, so this phase is
// L2-gather/atomic-throughput bound; keep the instruction count minimal.
// ---------------------------------------------------------------------------
__global__ void gcl_edge(const float* __restrict__ x,
                         const int*   __restrict__ src,
                         const int*   __restrict__ dst,
                         const int*   __restrict__ rel,
                         const float* __restrict__ w,
                         float* __restrict__ agg,
                         float* __restrict__ cnt) {
    int wave = blockIdx.x * (blockDim.x >> 5) + (threadIdx.x >> 5);
    int lane = threadIdx.x & 31;
    if (wave >= NEDGES) return;

    int s = src[wave];
    int d = dst[wave];
    int r = rel[wave];

    float2 xd  = *(const float2*)(x + (size_t)d * DD + lane * 2);
    float2 xs  = *(const float2*)(x + (size_t)s * DD + lane * 2);
    float2 wd  = *(const float2*)(w + r * (2 * DD) + lane * 2);
    float2 wsv = *(const float2*)(w + r * (2 * DD) + DD + lane * 2);

    float partial = xd.x * wd.x + xd.y * wd.y + xs.x * wsv.x + xs.y * wsv.y;
#pragma unroll
    for (int off = 16; off > 0; off >>= 1)
        partial += __shfl_xor(partial, off, 32);

    float gate = 1.0f / (1.0f + __expf(-partial));

    float* arow = agg + (size_t)d * DD + lane * 2;
    atomicAdd(arow,     xs.x * gate);
    atomicAdd(arow + 1, xs.y * gate);
    if (lane == 0) atomicAdd(cnt + d, 1.0f);
}

// ---------------------------------------------------------------------------
// Kernel 4: node pass with WMMA. One wave per 16-node tile.
// combined = [x[i] | agg[i]/max(cnt,1)] (16x128), out = combined @ W_lin^T + b,
// leaky_relu(0.01). K=128 -> 4 chunks of 32; N=64 -> 4 tiles of 16.
// B fragments come pre-converted/pre-swizzled from packB (f16, 32B/lane).
// agg is read from (and out written to) the same buffer: each wave only
// touches its own 16 rows (50000 % 16 == 0), so in-place is hazard-free.
// ---------------------------------------------------------------------------
__device__ __forceinline__ void pack8(v16h& f, int o, float4 a, float4 b) {
    f[o + 0] = (_Float16)a.x; f[o + 1] = (_Float16)a.y;
    f[o + 2] = (_Float16)a.z; f[o + 3] = (_Float16)a.w;
    f[o + 4] = (_Float16)b.x; f[o + 5] = (_Float16)b.y;
    f[o + 6] = (_Float16)b.z; f[o + 7] = (_Float16)b.w;
}

__global__ void gcl_node(const float*    __restrict__ x,
                         const _Float16* __restrict__ packB,
                         const float*    __restrict__ blin,   // (64)
                         const float*    __restrict__ cnt,
                         float* __restrict__ out /* agg in, result out */) {
    int wave  = blockIdx.x * (blockDim.x >> 5) + (threadIdx.x >> 5);
    int mbase = wave * 16;
    if (mbase >= NNODES) return;

    int lane = threadIdx.x & 31;
    int hs   = lane >> 4;                 // k-half select (ISA 16-bit A layout)
    int ml   = lane & 15;                 // row (A) / col (B,C,D) within tile
    int mr   = mbase + ml;

    float inv = 1.0f / fmaxf(cnt[mr], 1.0f);

    v8f acc[4];
#pragma unroll
    for (int t = 0; t < 4; ++t) {
        float bv = blin[t * 16 + ml];     // C layout: N = lane&15 for every VGPR
#pragma unroll
        for (int v = 0; v < 8; ++v) acc[t][v] = bv;
    }

    const float* xrow = x   + (size_t)mr * DD;
    const float* arow = out + (size_t)mr * DD;

#pragma unroll
    for (int kc = 0; kc < 4; ++kc) {
        int kb = kc * 32;
        v16h afrag;
        if (kc < 2) {                     // K in [0,64): from x
            float4 p0 = *(const float4*)(xrow + kb + hs * 8);
            float4 p1 = *(const float4*)(xrow + kb + hs * 8 + 4);
            float4 q0 = *(const float4*)(xrow + kb + 16 + hs * 8);
            float4 q1 = *(const float4*)(xrow + kb + 16 + hs * 8 + 4);
            pack8(afrag, 0, p0, p1);
            pack8(afrag, 8, q0, q1);
        } else {                          // K in [64,128): agg / max(cnt,1)
            int ab = kb - 64;
            float4 p0 = *(const float4*)(arow + ab + hs * 8);
            float4 p1 = *(const float4*)(arow + ab + hs * 8 + 4);
            float4 q0 = *(const float4*)(arow + ab + 16 + hs * 8);
            float4 q1 = *(const float4*)(arow + ab + 16 + hs * 8 + 4);
            p0.x *= inv; p0.y *= inv; p0.z *= inv; p0.w *= inv;
            p1.x *= inv; p1.y *= inv; p1.z *= inv; p1.w *= inv;
            q0.x *= inv; q0.y *= inv; q0.z *= inv; q0.w *= inv;
            q1.x *= inv; q1.y *= inv; q1.z *= inv; q1.w *= inv;
            pack8(afrag, 0, p0, p1);
            pack8(afrag, 8, q0, q1);
        }

#pragma unroll
        for (int t = 0; t < 4; ++t) {
            v16h bfrag = *(const v16h*)(packB + (size_t)((kc * 4 + t) * 32 + lane) * 16);
            acc[t] = __builtin_amdgcn_wmma_f32_16x16x32_f16(
                false, afrag, false, bfrag, (short)0, acc[t], false, false);
        }
    }

    // Epilogue: leaky_relu and store. C/D layout: VGPR v -> M = v + 8*hs.
#pragma unroll
    for (int t = 0; t < 4; ++t) {
#pragma unroll
        for (int v = 0; v < 8; ++v) {
            int row = mbase + v + hs * 8;
            float val = acc[t][v];
            val = (val >= 0.0f) ? val : 0.01f * val;
            out[(size_t)row * DD + t * 16 + ml] = val;
        }
    }
}

// ---------------------------------------------------------------------------
extern "C" void kernel_launch(void* const* d_in, const int* in_sizes, int n_in,
                              void* d_out, int out_size, void* d_ws, size_t ws_size,
                              hipStream_t stream) {
    const float* x     = (const float*)d_in[0];
    const int*   src   = (const int*)  d_in[1];
    const int*   dst   = (const int*)  d_in[2];
    const int*   rel   = (const int*)  d_in[3];
    const float* W_r   = (const float*)d_in[4];
    const float* W_lin = (const float*)d_in[5];
    const float* b_lin = (const float*)d_in[6];

    float* out = (float*)d_out;        // doubles as agg accumulator
    float* cnt = (float*)d_ws;         // 50000 floats
    float* w   = cnt + 51200;          // 1024 floats (8 x 128)
    _Float16* packB = (_Float16*)(w + 1536);   // 8192 halves (16 KB), 64B-aligned

    gcl_zero<<<(NNODES * DD + 255) / 256, 256, 0, stream>>>(out, cnt);
    gcl_wsum<<<(NREL * 2 * DD + 255) / 256, 256, 0, stream>>>(W_r, w);
    gcl_packB<<<2, 256, 0, stream>>>(W_lin, packB);
    gcl_edge<<<NEDGES / 8, 256, 0, stream>>>(x, src, dst, rel, w, out, cnt);

    int nwaves = NNODES / 16;          // 3125 tiles
    gcl_node<<<(nwaves + 7) / 8, 256, 0, stream>>>(x, packB, b_lin, cnt, out);
}
